// GNN_31748398252732
// MI455X (gfx1250) — compile-verified, matrix operand order checked by reading, not run on hardware
//
#include <hip/hip_runtime.h>
#include <hip/hip_bf16.h>

// ---------------------------------------------------------------------------
// GNN forward for MI455X (gfx1250, wave32).
//
// d_in layout (setup_inputs insertion order; nested 'params' flattened via
// jax pytree rules = alphabetical dict keys):
//   0:x_h 1:x_g 2:edge_attr 3:u 4:noise
//   5 + blk*30 + off, blk=0..3:
//     off: 0 bn_e.b  1 bn_e.g  2 bn_g.b  3 bn_g.g  4 bn_h.b  5 bn_h.g
//          6 edge.b1 7 edge.b2 8 edge.w1 9 edge.w2
//         10 g1.b1  11 g1.b2  12 g1.w1  13 g1.w2
//         14 g2.b1  15 g2.b2  16 g2.w1  17 g2.w2
//         18 glob.b1 19 glob.b2 20 glob.w1 21 glob.w2
//         22 h1.b1  23 h1.b2  24 h1.w1  25 h1.w2
//         26 h2.b1  27 h2.b2  28 h2.w1  29 h2.w2
//   125..128: edge_out b1,b2,w1,w2
//   129: edge_index (2 x E int32), 130..132: batch_* (unused, B=1)
// ---------------------------------------------------------------------------

#define DEVINL __device__ __forceinline__

typedef __attribute__((ext_vector_type(16))) _Float16 v16h;
typedef __attribute__((ext_vector_type(8)))  float    v8f;
typedef _Float16 f16;

static constexpr int N_Hc = 50000;
static constexpr int N_Gc = 50000;
static constexpr int Ec   = 800000;
static constexpr int TILES_E = Ec / 16;    // 50000 (exact)
static constexpr int TILES_H = N_Hc / 16;  // 3125  (exact)

// ---- WMMA operand helpers (f16 16x16x32, wave32) ---------------------------
// 16-bit A layout (ISA 7.12.2): lanes 0-15 hold K 0..7 (V0..3) and 16..23
// (V4..7); lanes 16-31 hold K 8..15 and 24..31.  B assumed mirrored with
// lane -> column N.  Both operands are therefore loaded with the SAME
// pattern from row-major [laneRow][K] LDS tiles: 2 contiguous 8-half runs
// per lane => ds_load_b128 pairs.  Row strides are padded (+8 halves) so the
// 16 row bases hit distinct bank groups (conflict-free b128 reads).
DEVINL int kmap(int h, int g) { return (h < 8) ? (h + 8 * g) : (8 + h + 8 * g); }

DEVINL v16h ldO(const f16* M, int ld) {  // M: 16 x (>=32) row-major tile
  const int l = threadIdx.x & 31, g = l >> 4, row = l & 15;
  v16h a;
#pragma unroll
  for (int h = 0; h < 16; ++h) a[h] = M[row * ld + kmap(h, g)];
  return a;
}

DEVINL v8f wmma16(v16h a, v16h b, v8f c) {
  return __builtin_amdgcn_wmma_f32_16x16x32_f16(false, a, false, b, (short)0, c,
                                                false, false);
}

// LeakyReLU(0.1)(x) == max(x, 0.1x): v_mul + v_max_num, no cmp/cndmask chain.
DEVINL float leaky01(float x) { return fmaxf(x, 0.1f * x); }

// Y(DOUT x 16) = W(DOUT x KPAD) * X(KPAD x 16) + bias, optional LeakyReLU(0.1).
// W: [DOUT][ldW] f16 LDS (cols >= KPAD valid, zero-padded K).
// X: [16][ldX] f16 LDS, lane-major (row = edge/node, col = K), zero-padded K.
template <int DOUT, int KPAD>
DEVINL void layer(const f16* W, int ldW, const float* Bias, const f16* X,
                  int ldX, v8f* acc, bool leaky) {
  const int l = threadIdx.x & 31, g = l >> 4;
#pragma unroll
  for (int mt = 0; mt < DOUT / 16; ++mt) {
    v8f c;
#pragma unroll
    for (int r = 0; r < 8; ++r) c[r] = Bias[mt * 16 + r + 8 * g];
#pragma unroll
    for (int kc = 0; kc < KPAD / 32; ++kc)
      c = wmma16(ldO(W + mt * 16 * ldW + kc * 32, ldW), ldO(X + kc * 32, ldX), c);
    if (leaky) {
#pragma unroll
      for (int r = 0; r < 8; ++r) c[r] = leaky01(c[r]);
    }
    acc[mt] = c;
  }
}

// Store C/D-layout f32 tiles as f16 cols mt*16+r+8g of lane-major staging.
// Per lane: 8 contiguous halves (16B aligned) -> single b128 LDS store.
template <int DOUT>
DEVINL void stageH(const v8f* acc, f16* X, int ldX) {
  const int l = threadIdx.x & 31, g = l >> 4, n = l & 15;
#pragma unroll
  for (int mt = 0; mt < DOUT / 16; ++mt)
#pragma unroll
    for (int r = 0; r < 8; ++r)
      X[n * ldX + mt * 16 + 8 * g + r] = (f16)acc[mt][r];
}

DEVINL float sigmoidf_(float x) { return 1.f / (1.f + expf(-x)); }

// ---- Pass 1: edge MLP (96->16->16) + h1 (48->48->48) + moment scatters -----
__global__ __launch_bounds__(128) void gnn_edge_h1(
    const float* __restrict__ xh, const float* __restrict__ xg,
    const float* __restrict__ ea_in, const float* __restrict__ uu,
    const int* __restrict__ src, const int* __restrict__ tgt,
    const float* __restrict__ We1, const float* __restrict__ Be1,
    const float* __restrict__ We2, const float* __restrict__ Be2,
    const float* __restrict__ Wh1, const float* __restrict__ Bh1,
    const float* __restrict__ Wh2, const float* __restrict__ Bh2,
    float* __restrict__ ea_out, float* __restrict__ cnt,
    float* __restrict__ s1, float* __restrict__ s2,
    float* __restrict__ bne_sum, float* __restrict__ bne_sq) {
  __shared__ f16 sWe1[16 * 104], sWe2[16 * 40], sWh1[48 * 72], sWh2[48 * 72];
  __shared__ float sB[128];
  __shared__ f16 sX[4][16 * 104], sH[4][16 * 40], sXh[4][16 * 72], sHh[4][16 * 72];
  __shared__ int sSrc[4][16];
  __shared__ float sStat[2][16];

  const int tid = threadIdx.x;
  for (int i = tid; i < 16 * 96; i += 128) sWe1[(i / 96) * 104 + (i % 96)] = (f16)We1[i];
  for (int i = tid; i < 16 * 32; i += 128) {
    int r = i >> 5, c = i & 31;
    sWe2[r * 40 + c] = (f16)((c < 16) ? We2[r * 16 + c] : 0.f);
  }
  for (int i = tid; i < 48 * 64; i += 128) {
    int r = i >> 6, c = i & 63;
    sWh1[r * 72 + c] = (f16)((c < 48) ? Wh1[r * 48 + c] : 0.f);
    sWh2[r * 72 + c] = (f16)((c < 48) ? Wh2[r * 48 + c] : 0.f);
  }
  if (tid < 16) { sB[tid] = Be1[tid]; sB[16 + tid] = Be2[tid]; }
  if (tid < 48) { sB[32 + tid] = Bh1[tid]; sB[80 + tid] = Bh2[tid]; }
  if (tid < 32) sStat[tid >> 4][tid & 15] = 0.f;
  const int w = tid >> 5, l = tid & 31, g = l >> 4, hn = l & 15;
  for (int i = l; i < 16 * 40; i += 32) sH[w][i] = (f16)0.f;
  for (int i = l; i < 16 * 72; i += 32) { sXh[w][i] = (f16)0.f; sHh[w][i] = (f16)0.f; }
  __syncthreads();

  const int nwaves = gridDim.x * 4;
  const int wave = blockIdx.x * 4 + w;
  const int niter = (TILES_E + nwaves - 1) / nwaves;
  for (int it = 0; it < niter; ++it) {
    const int tile = wave + it * nwaves;
    const bool active = tile < TILES_E;
    const int e = (active ? tile * 16 : 0) + hn;
    {
      const int is = src[e], itg = tgt[e];
      if (g == 0) {
        sSrc[w][hn] = is;
        for (int j = 0; j < 32; ++j) sX[w][hn * 104 + j] = (f16)xh[is * 32 + j];
        for (int j = 0; j < 16; ++j) sX[w][hn * 104 + 64 + j] = (f16)ea_in[e * 16 + j];
      } else {
        for (int j = 0; j < 32; ++j) {
          const float v = xg[itg * 32 + j];
          sX[w][hn * 104 + 32 + j] = (f16)v;
          sXh[w][hn * 72 + j] = (f16)v;
        }
        for (int j = 0; j < 16; ++j) sX[w][hn * 104 + 80 + j] = (f16)uu[j];
      }
    }
    __syncthreads();
    v8f acc[3];
    layer<16, 96>(sWe1, 104, sB, sX[w], 104, acc, true);
    stageH<16>(acc, sH[w], 40);
    __syncthreads();
    layer<16, 32>(sWe2, 40, sB + 16, sH[w], 40, acc, false);
#pragma unroll
    for (int r = 0; r < 8; ++r)  // feed h1 input (contiguous per lane)
      sXh[w][hn * 72 + 32 + 8 * g + r] = (f16)acc[0][r];
    if (active) {
#pragma unroll
      for (int r = 0; r < 8; ++r) {
        const int mm = 8 * g + r;
        const float v = acc[0][r];
        ea_out[e * 16 + mm] = v;
        atomicAdd(&sStat[0][mm], v);
        atomicAdd(&sStat[1][mm], v * v);
      }
    }
    __syncthreads();
    v8f a1[3];
    layer<48, 64>(sWh1, 72, sB + 32, sXh[w], 72, a1, true);
    stageH<48>(a1, sHh[w], 72);
    __syncthreads();
    layer<48, 64>(sWh2, 72, sB + 80, sHh[w], 72, a1, false);
    if (active) {
      const int s = sSrc[w][hn];
#pragma unroll
      for (int mt = 0; mt < 3; ++mt)
#pragma unroll
        for (int r = 0; r < 8; ++r) {
          const int mm = mt * 16 + 8 * g + r;
          const float v = a1[mt][r];
          atomicAdd(&s1[s * 48 + mm], v);
          atomicAdd(&s2[s * 48 + mm], v * v);
        }
      if (g == 0) atomicAdd(&cnt[s], 1.f);
    }
    __syncthreads();
  }
  if (tid < 16) {
    atomicAdd(&bne_sum[tid], sStat[0][tid]);
    atomicAdd(&bne_sq[tid], sStat[1][tid]);
  }
}

// ---- mean / std prep per (h-node, feature) --------------------------------
__global__ void gnn_momprep(const float* __restrict__ cnt,
                            const float* __restrict__ s1,
                            const float* __restrict__ s2,
                            float* __restrict__ amean, float* __restrict__ bstd) {
  const int i = blockIdx.x * blockDim.x + threadIdx.x;
  if (i >= N_Hc * 48) return;
  const float c = fmaxf(cnt[i / 48], 1.f);
  const float a = s1[i] / c;
  const float v = s2[i] / c - a * a;
  amean[i] = a;
  bstd[i] = sqrtf(1e-6f + fmaxf(v, 0.f));
}

// ---- Pass 2: recompute h1, scatter diff^3 / diff^4 -------------------------
__global__ __launch_bounds__(128) void gnn_mom_scatter(
    const float* __restrict__ xg, const float* __restrict__ ea,
    const int* __restrict__ src, const int* __restrict__ tgt,
    const float* __restrict__ Wh1, const float* __restrict__ Bh1,
    const float* __restrict__ Wh2, const float* __restrict__ Bh2,
    const float* __restrict__ amean, float* __restrict__ s3,
    float* __restrict__ s4) {
  __shared__ f16 sW1[48 * 72], sW2[48 * 72];
  __shared__ float sB[96];
  __shared__ f16 sXh[4][16 * 72], sHh[4][16 * 72];
  __shared__ int sSrc[4][16];
  const int tid = threadIdx.x;
  for (int i = tid; i < 48 * 64; i += 128) {
    int r = i >> 6, c = i & 63;
    sW1[r * 72 + c] = (f16)((c < 48) ? Wh1[r * 48 + c] : 0.f);
    sW2[r * 72 + c] = (f16)((c < 48) ? Wh2[r * 48 + c] : 0.f);
  }
  if (tid < 48) { sB[tid] = Bh1[tid]; sB[48 + tid] = Bh2[tid]; }
  const int w = tid >> 5, l = tid & 31, g = l >> 4, hn = l & 15;
  for (int i = l; i < 16 * 72; i += 32) { sXh[w][i] = (f16)0.f; sHh[w][i] = (f16)0.f; }
  __syncthreads();
  const int nwaves = gridDim.x * 4, wave = blockIdx.x * 4 + w;
  const int niter = (TILES_E + nwaves - 1) / nwaves;
  for (int it = 0; it < niter; ++it) {
    const int tile = wave + it * nwaves;
    const bool active = tile < TILES_E;
    const int e = (active ? tile * 16 : 0) + hn;
    {
      const int is = src[e], itg = tgt[e];
      if (g == 0) {
        sSrc[w][hn] = is;
        for (int j = 0; j < 32; ++j) sXh[w][hn * 72 + j] = (f16)xg[itg * 32 + j];
      } else {
        for (int j = 0; j < 16; ++j) sXh[w][hn * 72 + 32 + j] = (f16)ea[e * 16 + j];
      }
    }
    __syncthreads();
    v8f a1[3];
    layer<48, 64>(sW1, 72, sB, sXh[w], 72, a1, true);
    stageH<48>(a1, sHh[w], 72);
    __syncthreads();
    layer<48, 64>(sW2, 72, sB + 48, sHh[w], 72, a1, false);
    if (active) {
      const int s = sSrc[w][hn];
#pragma unroll
      for (int mt = 0; mt < 3; ++mt)
#pragma unroll
        for (int r = 0; r < 8; ++r) {
          const int mm = mt * 16 + 8 * g + r;
          const float df = a1[mt][r] - amean[s * 48 + mm];
          const float d3 = df * df * df;
          atomicAdd(&s3[s * 48 + mm], d3);
          atomicAdd(&s4[s * 48 + mm], d3 * df);
        }
    }
    __syncthreads();
  }
}

// ---- h2 per h-node (241 -> 32 -> 32) --------------------------------------
__global__ __launch_bounds__(128) void gnn_h2(
    const float* __restrict__ xh_in, const float* __restrict__ uu,
    const float* __restrict__ cnt, const float* __restrict__ amean,
    const float* __restrict__ bstd, const float* __restrict__ s3,
    const float* __restrict__ s4, const float* __restrict__ W1,
    const float* __restrict__ B1, const float* __restrict__ W2,
    const float* __restrict__ B2, float* __restrict__ xh_new,
    float* __restrict__ bsum, float* __restrict__ bsq) {
  __shared__ f16 sW1[32 * 264], sW2[32 * 40];
  __shared__ float sB[64];
  __shared__ f16 sX[4][16 * 264], sH[4][16 * 40];
  __shared__ float sStat[2][32];
  const int tid = threadIdx.x;
  for (int i = tid; i < 32 * 256; i += 128) {
    int r = i >> 8, c = i & 255;
    sW1[r * 264 + c] = (f16)((c < 241) ? W1[r * 241 + c] : 0.f);
  }
  for (int i = tid; i < 32 * 32; i += 128) {
    int r = i >> 5, c = i & 31;
    sW2[r * 40 + c] = (f16)W2[i];
  }
  if (tid < 32) { sB[tid] = B1[tid]; sB[32 + tid] = B2[tid]; sStat[0][tid] = 0.f; sStat[1][tid] = 0.f; }
  const int w = tid >> 5, l = tid & 31, g = l >> 4, hn = l & 15;
  for (int i = l; i < 16 * 264; i += 32) sX[w][i] = (f16)0.f;  // pads K 241..255
  __syncthreads();
  const int nwaves = gridDim.x * 4, wave = blockIdx.x * 4 + w;
  const int niter = (TILES_H + nwaves - 1) / nwaves;
  for (int it = 0; it < niter; ++it) {
    const int tile = wave + it * nwaves;
    const bool active = tile < TILES_H;
    const int nd = active ? tile * 16 + hn : 0;
    if (g == 0) {
      for (int j = 0; j < 32; ++j) sX[w][hn * 264 + j] = (f16)xh_in[nd * 32 + j];
      sX[w][hn * 264 + 32] = (f16)cnt[nd];  // raw n
      for (int j = 0; j < 48; ++j) sX[w][hn * 264 + 33 + j] = (f16)amean[nd * 48 + j];
    } else {
      const float cc = fmaxf(cnt[nd], 1.f);
      for (int j = 0; j < 48; ++j) {
        const float bb = bstd[nd * 48 + j];
        const float b3 = bb * bb * bb;
        sX[w][hn * 264 + 81 + j] = (f16)bb;
        sX[w][hn * 264 + 129 + j] = (f16)((s3[nd * 48 + j] / cc) / b3);
        sX[w][hn * 264 + 177 + j] = (f16)((s4[nd * 48 + j] / cc) / (b3 * bb));
      }
      for (int j = 0; j < 16; ++j) sX[w][hn * 264 + 225 + j] = (f16)uu[j];
    }
    __syncthreads();
    v8f acc[2];
    layer<32, 256>(sW1, 264, sB, sX[w], 264, acc, true);
    stageH<32>(acc, sH[w], 40);
    __syncthreads();
    layer<32, 32>(sW2, 40, sB + 32, sH[w], 40, acc, false);
    if (active) {
#pragma unroll
      for (int mt = 0; mt < 2; ++mt)
#pragma unroll
        for (int r = 0; r < 8; ++r) {
          const int mm = mt * 16 + 8 * g + r;
          const float v = acc[mt][r];
          xh_new[nd * 32 + mm] = v;
          atomicAdd(&sStat[0][mm], v);
          atomicAdd(&sStat[1][mm], v * v);
        }
    }
    __syncthreads();
  }
  if (tid < 32) { atomicAdd(&bsum[tid], sStat[0][tid]); atomicAdd(&bsq[tid], sStat[1][tid]); }
}

// ---- g1 per edge (48->48->48) scatter to agg[tgt] -------------------------
__global__ __launch_bounds__(128) void gnn_g1(
    const float* __restrict__ xh, const float* __restrict__ ea,
    const int* __restrict__ src, const int* __restrict__ tgt,
    const float* __restrict__ W1, const float* __restrict__ B1,
    const float* __restrict__ W2, const float* __restrict__ B2,
    float* __restrict__ agg) {
  __shared__ f16 sW1[48 * 72], sW2[48 * 72];
  __shared__ float sB[96];
  __shared__ f16 sXh[4][16 * 72], sHh[4][16 * 72];
  __shared__ int sTgt[4][16];
  const int tid = threadIdx.x;
  for (int i = tid; i < 48 * 64; i += 128) {
    int r = i >> 6, c = i & 63;
    sW1[r * 72 + c] = (f16)((c < 48) ? W1[r * 48 + c] : 0.f);
    sW2[r * 72 + c] = (f16)((c < 48) ? W2[r * 48 + c] : 0.f);
  }
  if (tid < 48) { sB[tid] = B1[tid]; sB[48 + tid] = B2[tid]; }
  const int w = tid >> 5, l = tid & 31, g = l >> 4, hn = l & 15;
  for (int i = l; i < 16 * 72; i += 32) { sXh[w][i] = (f16)0.f; sHh[w][i] = (f16)0.f; }
  __syncthreads();
  const int nwaves = gridDim.x * 4, wave = blockIdx.x * 4 + w;
  const int niter = (TILES_E + nwaves - 1) / nwaves;
  for (int it = 0; it < niter; ++it) {
    const int tile = wave + it * nwaves;
    const bool active = tile < TILES_E;
    const int e = (active ? tile * 16 : 0) + hn;
    {
      const int is = src[e], itg = tgt[e];
      if (g == 0) {
        sTgt[w][hn] = itg;
        for (int j = 0; j < 32; ++j) sXh[w][hn * 72 + j] = (f16)xh[is * 32 + j];
      } else {
        for (int j = 0; j < 16; ++j) sXh[w][hn * 72 + 32 + j] = (f16)ea[e * 16 + j];
      }
    }
    __syncthreads();
    v8f a1[3];
    layer<48, 64>(sW1, 72, sB, sXh[w], 72, a1, true);
    stageH<48>(a1, sHh[w], 72);
    __syncthreads();
    layer<48, 64>(sW2, 72, sB + 48, sHh[w], 72, a1, false);
    if (active) {
      const int t = sTgt[w][hn];
#pragma unroll
      for (int mt = 0; mt < 3; ++mt)
#pragma unroll
        for (int r = 0; r < 8; ++r)
          atomicAdd(&agg[t * 48 + mt * 16 + 8 * g + r], a1[mt][r]);
    }
    __syncthreads();
  }
}

// ---- g2 per g-node (96 -> 32 -> 32) ---------------------------------------
__global__ __launch_bounds__(128) void gnn_g2(
    const float* __restrict__ xg_in, const float* __restrict__ agg,
    const float* __restrict__ uu, const float* __restrict__ W1,
    const float* __restrict__ B1, const float* __restrict__ W2,
    const float* __restrict__ B2, float* __restrict__ xg_new,
    float* __restrict__ bsum, float* __restrict__ bsq) {
  __shared__ f16 sW1[32 * 104], sW2[32 * 40];
  __shared__ float sB[64];
  __shared__ f16 sX[4][16 * 104], sH[4][16 * 40];
  __shared__ float sStat[2][32];
  const int tid = threadIdx.x;
  for (int i = tid; i < 32 * 96; i += 128) sW1[(i / 96) * 104 + (i % 96)] = (f16)W1[i];
  for (int i = tid; i < 32 * 32; i += 128) {
    int r = i >> 5, c = i & 31;
    sW2[r * 40 + c] = (f16)W2[i];
  }
  if (tid < 32) { sB[tid] = B1[tid]; sB[32 + tid] = B2[tid]; sStat[0][tid] = 0.f; sStat[1][tid] = 0.f; }
  const int w = tid >> 5, l = tid & 31, g = l >> 4, hn = l & 15;
  __syncthreads();
  const int nwaves = gridDim.x * 4, wave = blockIdx.x * 4 + w;
  const int niter = (TILES_H + nwaves - 1) / nwaves;
  for (int it = 0; it < niter; ++it) {
    const int tile = wave + it * nwaves;
    const bool active = tile < TILES_H;
    const int nd = active ? tile * 16 + hn : 0;
    if (g == 0) {
      for (int j = 0; j < 32; ++j) sX[w][hn * 104 + j] = (f16)xg_in[nd * 32 + j];
      for (int j = 0; j < 16; ++j) sX[w][hn * 104 + 80 + j] = (f16)uu[j];
    } else {
      for (int j = 0; j < 48; ++j) sX[w][hn * 104 + 32 + j] = (f16)agg[nd * 48 + j];
    }
    __syncthreads();
    v8f acc[2];
    layer<32, 96>(sW1, 104, sB, sX[w], 104, acc, true);
    stageH<32>(acc, sH[w], 40);
    __syncthreads();
    layer<32, 32>(sW2, 40, sB + 32, sH[w], 40, acc, false);
    if (active) {
#pragma unroll
      for (int mt = 0; mt < 2; ++mt)
#pragma unroll
        for (int r = 0; r < 8; ++r) {
          const int mm = mt * 16 + 8 * g + r;
          const float v = acc[mt][r];
          xg_new[nd * 32 + mm] = v;
          atomicAdd(&sStat[0][mm], v);
          atomicAdd(&sStat[1][mm], v * v);
        }
    }
    __syncthreads();
  }
  if (tid < 32) { atomicAdd(&bsum[tid], sStat[0][tid]); atomicAdd(&bsq[tid], sStat[1][tid]); }
}

// ---- global MLP (80 -> 16 -> 16), single wave -----------------------------
__global__ void gnn_glob(const float* __restrict__ u_in,
                         const float* __restrict__ hsum,
                         const float* __restrict__ gsum,
                         const float* __restrict__ W1, const float* __restrict__ B1,
                         const float* __restrict__ W2, const float* __restrict__ B2,
                         float* __restrict__ u_out) {
  __shared__ float xin[80], hh[16];
  const int t = threadIdx.x;
  if (t < 16) xin[t] = u_in[t];
  if (t < 32) { xin[16 + t] = hsum[t] * (1.f / N_Hc); xin[48 + t] = gsum[t] * (1.f / N_Gc); }
  __syncthreads();
  if (t < 16) {
    float a = B1[t];
    for (int k = 0; k < 80; ++k) a += W1[t * 80 + k] * xin[k];
    hh[t] = fmaxf(a, 0.1f * a);
  }
  __syncthreads();
  if (t < 16) {
    float a = B2[t];
    for (int k = 0; k < 16; ++k) a += W2[t * 16 + k] * hh[k];
    u_out[t] = a;
  }
}

// ---- BatchNorm apply (biased batch stats) ---------------------------------
__global__ void gnn_bn(const float* __restrict__ x, float* __restrict__ y,
                       const float* __restrict__ sum, const float* __restrict__ sq,
                       const float* __restrict__ gamma, const float* __restrict__ beta,
                       int N, int F) {
  const long i = (long)blockIdx.x * blockDim.x + threadIdx.x;
  if (i >= (long)N * F) return;
  const int f = (int)(i % F);
  const float inv = 1.f / (float)N;
  const float m = sum[f] * inv;
  const float v = sq[f] * inv - m * m;
  y[i] = (x[i] - m) * rsqrtf(v + 1e-5f) * gamma[f] + beta[f];
}

// ---- final edge head (96 -> 1 -> 1) + time transform ----------------------
__global__ void gnn_final(const float* __restrict__ xh, const float* __restrict__ xg,
                          const float* __restrict__ ea, const float* __restrict__ uu,
                          const float* __restrict__ noise,
                          const int* __restrict__ src, const int* __restrict__ tgt,
                          const float* __restrict__ W1, const float* __restrict__ B1,
                          const float* __restrict__ W2, const float* __restrict__ B2,
                          float* __restrict__ out) {
  __shared__ float sW[96], sS[3];
  if (threadIdx.x < 96) sW[threadIdx.x] = W1[threadIdx.x];
  if (threadIdx.x == 0) { sS[0] = B1[0]; sS[1] = W2[0]; sS[2] = B2[0]; }
  __syncthreads();
  const int e = blockIdx.x * blockDim.x + threadIdx.x;
  if (e >= Ec) return;
  const int is = src[e], itg = tgt[e];
  float a = sS[0];
  for (int j = 0; j < 32; ++j) a += sW[j] * xh[is * 32 + j];
  for (int j = 0; j < 32; ++j) a += sW[32 + j] * xg[itg * 32 + j];
  for (int j = 0; j < 16; ++j) a += sW[64 + j] * ea[e * 16 + j];
  for (int j = 0; j < 16; ++j) a += sW[80 + j] * uu[j];
  const float h = fmaxf(a, 0.1f * a);
  const float y = sS[1] * h + sS[2];
  float t = 100.f * sigmoidf_(y) + noise[e];
  const float it = floorf(t);
  out[e] = it + sigmoidf_(20.f * (t - 0.5f - it));
}

// ---------------------------------------------------------------------------
extern "C" void kernel_launch(void* const* d_in, const int* in_sizes, int n_in,
                              void* d_out, int out_size, void* d_ws, size_t ws_size,
                              hipStream_t stream) {
  const float* x_h = (const float*)d_in[0];
  const float* x_g = (const float*)d_in[1];
  const float* ea0 = (const float*)d_in[2];
  const float* u0 = (const float*)d_in[3];
  const float* noise = (const float*)d_in[4];
  const int* ei = (const int*)d_in[129];
  const int* src = ei;
  const int* tgt = ei + Ec;

  enum { BNE_B = 0, BNE_G, BNG_B, BNG_G, BNH_B, BNH_G,
         EDGE_B1, EDGE_B2, EDGE_W1, EDGE_W2,
         G1_B1, G1_B2, G1_W1, G1_W2,
         G2_B1, G2_B2, G2_W1, G2_W2,
         GL_B1, GL_B2, GL_W1, GL_W2,
         H1_B1, H1_B2, H1_W1, H1_W2,
         H2_B1, H2_B2, H2_W1, H2_W2 };
  auto P = [&](int blk, int off) { return (const float*)d_in[5 + blk * 30 + off]; };
  const float* EO_B1 = (const float*)d_in[125];
  const float* EO_B2 = (const float*)d_in[126];
  const float* EO_W1 = (const float*)d_in[127];
  const float* EO_W2 = (const float*)d_in[128];

  // ---- workspace carve (f32) ----
  float* p = (float*)d_ws;
  auto take = [&](size_t n) { float* r = p; p += n; return r; };
  float* ea_cur = take((size_t)Ec * 16);
  float* xh_cur = take((size_t)N_Hc * 32);
  float* xg_cur = take((size_t)N_Gc * 32);
  float* xh_new = take((size_t)N_Hc * 32);
  float* xg_new = take((size_t)N_Gc * 32);
  float* u_a = take(16);
  float* u_b = take(16);
  float* acc0 = p;  // contiguous region zeroed per block
  float* cnt = take(N_Hc);
  float* s1 = take((size_t)N_Hc * 48);
  float* s2 = take((size_t)N_Hc * 48);
  float* s3 = take((size_t)N_Hc * 48);
  float* s4 = take((size_t)N_Hc * 48);
  float* agg = take((size_t)N_Gc * 48);
  float* bnh_sum = take(32); float* bnh_sq = take(32);
  float* bng_sum = take(32); float* bng_sq = take(32);
  float* bne_sum = take(16); float* bne_sq = take(16);
  const size_t accBytes = (size_t)(p - acc0) * sizeof(float);
  float* amean = take((size_t)N_Hc * 48);
  float* bstd = take((size_t)N_Hc * 48);
  if ((size_t)(p - (float*)d_ws) * sizeof(float) > ws_size) return;  // ws too small

  const float* xh_in = x_h;
  const float* xg_in = x_g;
  const float* ea_in = ea0;
  const float* u_in = u0;
  float* u_ping = u_a;

  constexpr int GE = 2048;  // edge-pass grid (4 waves/block)
  constexpr int GN = 512;   // node-pass grid

  for (int b = 0; b < 4; ++b) {
    hipMemsetAsync(acc0, 0, accBytes, stream);
    gnn_edge_h1<<<GE, 128, 0, stream>>>(
        xh_in, xg_in, ea_in, u_in, src, tgt,
        P(b, EDGE_W1), P(b, EDGE_B1), P(b, EDGE_W2), P(b, EDGE_B2),
        P(b, H1_W1), P(b, H1_B1), P(b, H1_W2), P(b, H1_B2),
        ea_cur, cnt, s1, s2, bne_sum, bne_sq);
    gnn_momprep<<<(N_Hc * 48 + 255) / 256, 256, 0, stream>>>(cnt, s1, s2, amean, bstd);
    gnn_mom_scatter<<<GE, 128, 0, stream>>>(
        xg_in, ea_cur, src, tgt,
        P(b, H1_W1), P(b, H1_B1), P(b, H1_W2), P(b, H1_B2), amean, s3, s4);
    gnn_h2<<<GN, 128, 0, stream>>>(
        xh_in, u_in, cnt, amean, bstd, s3, s4,
        P(b, H2_W1), P(b, H2_B1), P(b, H2_W2), P(b, H2_B2),
        xh_new, bnh_sum, bnh_sq);
    gnn_g1<<<GE, 128, 0, stream>>>(
        xh_new, ea_cur, src, tgt,
        P(b, G1_W1), P(b, G1_B1), P(b, G1_W2), P(b, G1_B2), agg);
    gnn_g2<<<GN, 128, 0, stream>>>(
        xg_in, agg, u_in,
        P(b, G2_W1), P(b, G2_B1), P(b, G2_W2), P(b, G2_B2),
        xg_new, bng_sum, bng_sq);
    gnn_glob<<<1, 32, 0, stream>>>(u_in, bnh_sum, bng_sum,
                                   P(b, GL_W1), P(b, GL_B1), P(b, GL_W2), P(b, GL_B2),
                                   u_ping);
    gnn_bn<<<(N_Hc * 32 + 255) / 256, 256, 0, stream>>>(
        xh_new, xh_cur, bnh_sum, bnh_sq, P(b, BNH_G), P(b, BNH_B), N_Hc, 32);
    gnn_bn<<<(N_Gc * 32 + 255) / 256, 256, 0, stream>>>(
        xg_new, xg_cur, bng_sum, bng_sq, P(b, BNG_G), P(b, BNG_B), N_Gc, 32);
    gnn_bn<<<(Ec * 16 + 255) / 256, 256, 0, stream>>>(
        ea_cur, ea_cur, bne_sum, bne_sq, P(b, BNE_G), P(b, BNE_B), Ec, 16);
    xh_in = xh_cur; xg_in = xg_cur; ea_in = ea_cur;
    u_in = u_ping; u_ping = (u_ping == u_a) ? u_b : u_a;
  }

  gnn_final<<<(Ec + 255) / 256, 256, 0, stream>>>(
      xh_in, xg_in, ea_in, u_in, noise, src, tgt,
      EO_W1, EO_B1, EO_W2, EO_B2, (float*)d_out);
}
